// SwitchFeedForward_56315611185980
// MI455X (gfx1250) — matrix operand order repaired and checked by reading, not compiled, not run on hardware
//
#include <hip/hip_runtime.h>

#define DM 1024
#define DFF 4096
#define NE 8
#define NTOK 8192
#define MTILE 128
#define NTILE 128
#define KC 64

typedef __attribute__((ext_vector_type(16))) __bf16 v16bf;
typedef __attribute__((ext_vector_type(8)))  float  v8f;
typedef __attribute__((ext_vector_type(4)))  unsigned int v4u;

union Frag { v16bf v; v4u u[2]; };

#if __has_builtin(__builtin_amdgcn_s_wait_asynccnt)
#define WAIT_ASYNCCNT(n) __builtin_amdgcn_s_wait_asynccnt(n)
#else
#define WAIT_ASYNCCNT(n) asm volatile("s_wait_asynccnt " #n ::: "memory")
#endif

// One lane-granular async copy: 16B from global -> LDS, tracked by ASYNCcnt.
__device__ __forceinline__ void async16(unsigned lds_off, const __bf16* g) {
  asm volatile("global_load_async_to_lds_b128 %0, %1, off"
               :
               : "v"(lds_off), "v"(g)
               : "memory");
}

__global__ void zero_counters(int* counters) {
  if (threadIdx.x < NE) counters[threadIdx.x] = 0;
}

__global__ void cvt_bf16(const float* __restrict__ src, __bf16* __restrict__ dst, long n) {
  long i = (long)blockIdx.x * blockDim.x + threadIdx.x;
  long stride = (long)gridDim.x * blockDim.x;
  for (; i < n; i += stride) dst[i] = (__bf16)src[i];
}

// One wave per token: logits = x @ sw^T + sb, softmax, top-1 route.
__global__ __launch_bounds__(128) void gate_kernel(
    const float* __restrict__ x, const float* __restrict__ sw, const float* __restrict__ sb,
    float* __restrict__ probs, float* __restrict__ pmax_out,
    int* __restrict__ counters, int* __restrict__ idxlist) {
  __shared__ float ssw[NE * DM];
  for (int i = threadIdx.x; i < NE * DM; i += 128) ssw[i] = sw[i];
  __syncthreads();
  const int wv = threadIdx.x >> 5, lane = threadIdx.x & 31;
  const int tok = blockIdx.x * 4 + wv;
  const float* xr = x + (size_t)tok * DM;
  float acc[NE];
#pragma unroll
  for (int e = 0; e < NE; ++e) acc[e] = 0.f;
  for (int k = lane; k < DM; k += 32) {
    float xv = xr[k];
#pragma unroll
    for (int e = 0; e < NE; ++e) acc[e] += xv * ssw[e * DM + k];
  }
#pragma unroll
  for (int off = 16; off > 0; off >>= 1) {
#pragma unroll
    for (int e = 0; e < NE; ++e) acc[e] += __shfl_xor(acc[e], off, 32);
  }
  if (lane == 0) {
    float l[NE];
    float mx = -3.4e38f; int am = 0;
#pragma unroll
    for (int e = 0; e < NE; ++e) {
      l[e] = acc[e] + sb[e];
      if (l[e] > mx) { mx = l[e]; am = e; }   // first-occurrence argmax
    }
    float s = 0.f;
#pragma unroll
    for (int e = 0; e < NE; ++e) { l[e] = __expf(l[e] - mx); s += l[e]; }
    float inv = 1.f / s;
#pragma unroll
    for (int e = 0; e < NE; ++e) probs[tok * NE + e] = l[e] * inv;
    pmax_out[tok] = l[am] * inv;
    int slot = atomicAdd(&counters[am], 1);
    idxlist[am * NTOK + slot] = tok;
  }
}

// Deterministic reductions for the small outputs.
__global__ void finalize_kernel(const float* __restrict__ probs, const int* __restrict__ counters,
                                float* __restrict__ out_counts, float* __restrict__ out_psum,
                                float* __restrict__ out_zero) {
  int t = threadIdx.x;
  if (t < NE) {
    float s = 0.f;
    for (int i = 0; i < NTOK; ++i) s += probs[i * NE + t];
    out_psum[t] = s;
    out_counts[t] = (float)counters[t];
  }
  if (t == NE) out_zero[0] = 0.f;
}

// Grouped GEMM with token gather/scatter, double-buffered async global->LDS pipeline.
// MODE 0: h[tok] = relu(x[tok] @ w1[e]^T + b1[e])   (bf16 out,  K=1024, N=4096)
// MODE 1: final[tok] = h[tok] @ w2[e]^T + b2[e]     (fp32 out,  K=4096, N=1024)
template <int MODE, int K, int N>
__global__ __launch_bounds__(256) void ffn_gemm(
    const __bf16* __restrict__ Abase, const __bf16* __restrict__ Bbase,
    const float* __restrict__ biasBase, void* __restrict__ outBase,
    const int* __restrict__ counters, const int* __restrict__ idxlist) {
  constexpr int LDSS = KC + 8;                  // 72 bf16 row stride (16B-aligned, bank-skewed)
  constexpr int BUFE = (MTILE + NTILE) * LDSS;  // elems per buffer: A tile then B tile
  constexpr unsigned BUFB = BUFE * 2;           // bytes per buffer
  constexpr int NCH = K / KC;

  extern __shared__ char smem[];                // 2*BUFB + MTILE*4 bytes
  __bf16* sBuf = (__bf16*)smem;
  int* sTok = (int*)(smem + 2 * BUFB);

  const int e = blockIdx.x >> 6;      // 64 worst-case m-blocks per expert
  const int mblk = blockIdx.x & 63;
  const int cnt = counters[e];
  const int m0 = mblk * MTILE;
  if (m0 >= cnt) return;              // data-dependent early exit (deterministic work)
  const int n0 = blockIdx.y * NTILE;

  const int tid = threadIdx.x;
  if (tid < MTILE) {
    int g = m0 + tid;
    sTok[tid] = idxlist[e * NTOK + (g < cnt ? g : 0)];  // clamp padding rows
  }
  __syncthreads();

  const __bf16* Bexp = Bbase + (size_t)e * N * K;

  // Per-thread copy plan: 4x16B of A (gathered token rows) + 4x16B of B per chunk.
  const __bf16* gsrc[8];
  unsigned loff[8];
#pragma unroll
  for (int j = 0; j < 4; ++j) {
    int q = j * 256 + tid;            // 0..1023
    int row = q >> 3, c8 = q & 7;     // 128 rows x 8 chunks of 8 bf16
    gsrc[j] = Abase + (size_t)sTok[row] * K + c8 * 8;
    loff[j] = (unsigned)(size_t)(sBuf + row * LDSS + c8 * 8);
    gsrc[4 + j] = Bexp + (size_t)(n0 + row) * K + c8 * 8;
    loff[4 + j] = (unsigned)(size_t)(sBuf + (MTILE + row) * LDSS + c8 * 8);
  }

  const int lane = tid & 31;
  const int wv = tid >> 5;
  const int R = (wv >> 1) * 32;   // wave row base (4 wave-rows)
  const int C = (wv & 1) * 64;    // wave col base (2 wave-cols)
  const int kh  = lane >> 4;      // K-half select within fragments
  const int c16 = lane & 15;

  v8f acc[2][4] = {};

  // Prologue: start chunk 0 into buffer 0.
#pragma unroll
  for (int j = 0; j < 8; ++j) async16(loff[j], gsrc[j]);

  for (int i = 0; i < NCH; ++i) {
    const int b = i & 1;
    if (i + 1 < NCH) {
      const unsigned nb = (1 - b) * BUFB;
#pragma unroll
      for (int j = 0; j < 8; ++j)
        async16(loff[j] + nb, gsrc[j] + (size_t)(i + 1) * KC);
      WAIT_ASYNCCNT(8);   // drain chunk i's 8 copies (in-order completion)
    } else {
      WAIT_ASYNCCNT(0);
    }
    __syncthreads();

    const __bf16* sA = sBuf + b * BUFE;
    const __bf16* sB = sA + MTILE * LDSS;
#pragma unroll
    for (int kstep = 0; kstep < 2; ++kstep) {
      const int ks = kstep * 32;
      Frag fa[2], fb[4];
#pragma unroll
      for (int m = 0; m < 2; ++m) {
        const __bf16* pa = sA + (R + m * 16 + c16) * LDSS + ks + kh * 8;
        fa[m].u[0] = *(const v4u*)pa;         // K = ks+kh*8 .. +7
        fa[m].u[1] = *(const v4u*)(pa + 16);  // K = ks+kh*8+16 .. +7
      }
#pragma unroll
      for (int n = 0; n < 4; ++n) {
        const __bf16* pb = sB + (C + n * 16 + c16) * LDSS + ks + kh * 8;
        fb[n].u[0] = *(const v4u*)pb;         // K = ks+kh*8 .. +7
        fb[n].u[1] = *(const v4u*)(pb + 16);  // K = ks+kh*8+16 .. +7
      }
#pragma unroll
      for (int m = 0; m < 2; ++m)
#pragma unroll
        for (int n = 0; n < 4; ++n)
          acc[m][n] = __builtin_amdgcn_wmma_f32_16x16x32_bf16(
              false, fa[m].v, false, fb[n].v, (short)0, acc[m][n], false, false);
    }
    __syncthreads();  // all waves done reading buf[b] before it is refilled
  }

  // Epilogue: bias (+relu for MODE 0), scatter by token id. D layout: lane=col,
  // VGPR i = row i (lanes 0-15) / row i+8 (lanes 16-31).
#pragma unroll
  for (int n = 0; n < 4; ++n) {
    const int ncol = n0 + C + n * 16 + c16;
    const float bias = biasBase[e * N + ncol];
#pragma unroll
    for (int m = 0; m < 2; ++m) {
#pragma unroll
      for (int i = 0; i < 8; ++i) {
        int mloc = R + m * 16 + kh * 8 + i;
        if (m0 + mloc < cnt) {
          float v = acc[m][n][i] + bias;
          int tok = sTok[mloc];
          if (MODE == 0) {
            v = fmaxf(v, 0.f);
            ((__bf16*)outBase)[(size_t)tok * N + ncol] = (__bf16)v;
          } else {
            ((float*)outBase)[(size_t)tok * N + ncol] = v;
          }
        }
      }
    }
  }
}

extern "C" void kernel_launch(void* const* d_in, const int* in_sizes, int n_in,
                              void* d_out, int out_size, void* d_ws, size_t ws_size,
                              hipStream_t stream) {
  const float* x  = (const float*)d_in[0];
  const float* sw = (const float*)d_in[1];
  const float* sb = (const float*)d_in[2];
  const float* w1 = (const float*)d_in[3];
  const float* b1 = (const float*)d_in[4];
  const float* w2 = (const float*)d_in[5];
  const float* b2 = (const float*)d_in[6];

  float* out = (float*)d_out;
  float* out_final  = out;                              // 8192*1024
  float* out_counts = out + (size_t)NTOK * DM;          // 8
  float* out_psum   = out_counts + NE;                  // 8
  float* out_zero   = out_psum + NE;                    // 1
  float* out_pmax   = out_zero + 1;                     // 8192

  char* ws = (char*)d_ws;
  size_t off = 0;
  int* counters = (int*)(ws + off);   off += 256;
  int* idxlist  = (int*)(ws + off);   off += (size_t)NE * NTOK * 4;
  float* probs  = (float*)(ws + off); off += (size_t)NTOK * NE * 4;
  __bf16* xb  = (__bf16*)(ws + off);  off += (size_t)NTOK * DM * 2;
  __bf16* w1b = (__bf16*)(ws + off);  off += (size_t)NE * DFF * DM * 2;
  __bf16* w2b = (__bf16*)(ws + off);  off += (size_t)NE * DM * DFF * 2;
  __bf16* h   = (__bf16*)(ws + off);  off += (size_t)NTOK * DFF * 2;   // ~209 MB total

  const size_t smem_bytes = (size_t)2 * (MTILE + NTILE) * (KC + 8) * 2 + MTILE * 4;

  zero_counters<<<1, 32, 0, stream>>>(counters);
  gate_kernel<<<NTOK / 4, 128, 0, stream>>>(x, sw, sb, probs, out_pmax, counters, idxlist);
  cvt_bf16<<<4096, 256, 0, stream>>>(x,  xb,  (long)NTOK * DM);
  cvt_bf16<<<8192, 256, 0, stream>>>(w1, w1b, (long)NE * DFF * DM);
  cvt_bf16<<<8192, 256, 0, stream>>>(w2, w2b, (long)NE * DM * DFF);
  finalize_kernel<<<1, 32, 0, stream>>>(probs, counters, out_counts, out_psum, out_zero);
  ffn_gemm<0, DM, DFF><<<dim3(NE * 64, DFF / NTILE), 256, smem_bytes, stream>>>(
      xb, w1b, b1, (void*)h, counters, idxlist);
  ffn_gemm<1, DFF, DM><<<dim3(NE * 64, DM / NTILE), 256, smem_bytes, stream>>>(
      h, w2b, b2, (void*)out_final, counters, idxlist);
}
